// ARGenerator_66812511257202
// MI455X (gfx1250) — matrix operand order repaired and checked by reading, not compiled, not run on hardware
//
#include <hip/hip_runtime.h>
#include <hip/hip_bf16.h>
#include <stdint.h>

// Problem sizes (fixed by the reference)
#define B_SZ  8192
#define S_IN  4096
#define S_OUT 4096
#define H_SZ  128
#define AR_N  7

typedef __attribute__((ext_vector_type(16))) __bf16 v16bf;
typedef __attribute__((ext_vector_type(8)))  float  v8f;

// 32-byte fragment shared between bf16-vector view (WMMA operand) and raw views.
union ABFrag {
  v16bf    v;
  uint32_t u[8];
  uint4    q[2];
};

// Pack two f32 into packed bf16 (round-to-nearest, ties away): 2 adds + 1 v_perm_b32.
__device__ __forceinline__ uint32_t pk_bf16(float lo, float hi) {
  uint32_t a = __float_as_uint(lo) + 0x8000u;
  uint32_t b = __float_as_uint(hi) + 0x8000u;
  // D[15:0] = a[31:16], D[31:16] = b[31:16]
  return __builtin_amdgcn_perm(b, a, 0x07060302u);
}

__device__ __forceinline__ uint16_t f32_to_bf16(float f) {
  return (uint16_t)((__float_as_uint(f) + 0x8000u) >> 16);
}

__device__ __forceinline__ v8f vzero8() {
  v8f z;
#pragma unroll
  for (int i = 0; i < 8; ++i) z[i] = 0.f;
  return z;
}

__device__ __forceinline__ float fast_tanhf(float x) {
  // 1 - 2/(e^{2x}+1); saturates correctly to +/-1
  return 1.f - 2.f / (__expf(2.f * x) + 1.f);
}

__device__ __forceinline__ void pack_a(ABFrag& A, const float4* r) {
  A.u[0] = pk_bf16(r[0].x, r[0].y); A.u[1] = pk_bf16(r[0].z, r[0].w);
  A.u[2] = pk_bf16(r[1].x, r[1].y); A.u[3] = pk_bf16(r[1].z, r[1].w);
  A.u[4] = pk_bf16(r[2].x, r[2].y); A.u[5] = pk_bf16(r[2].z, r[2].w);
  A.u[6] = pk_bf16(r[3].x, r[3].y); A.u[7] = pk_bf16(r[3].z, r[3].w);
}

// ---------------------------------------------------------------------------
// f32 -> bf16 conversion (weights, run once per call; tiny)
// ---------------------------------------------------------------------------
__global__ void __launch_bounds__(256) cvt_bf16_kernel(const float* __restrict__ src,
                                                       uint16_t* __restrict__ dst, int n) {
  int i = blockIdx.x * 256 + threadIdx.x;
  if (i < n) dst[i] = f32_to_bf16(src[i]);
}

// ---------------------------------------------------------------------------
// AR(7) recurrence: out[b,t] = noise[b,t] + sum_i c[i]*out[b,t-1-i] (t>=7), 0 else.
// One lane per batch row. Critical path reduced to ONE fma per step:
//   s_t uses only y_{t-2..t-7}; y_t = fma(c0, y_{t-1}, s_t).
// ---------------------------------------------------------------------------
__global__ void __launch_bounds__(256) ar_filter_kernel(const float* __restrict__ noise,
                                                        const float* __restrict__ coef,
                                                        float* __restrict__ out) {
  const int b = blockIdx.x * 256 + threadIdx.x;
  const float* nrow = noise + (size_t)b * S_OUT;
  float* orow = out + (size_t)b * S_OUT;

  const float c0 = coef[0], c1 = coef[1], c2 = coef[2], c3 = coef[3],
              c4 = coef[4], c5 = coef[5], c6 = coef[6];
  float y1 = 0.f, y2 = 0.f, y3 = 0.f, y4 = 0.f, y5 = 0.f, y6 = 0.f, y7 = 0.f;

#pragma unroll
  for (int t = 0; t < AR_N; ++t) orow[t] = 0.f;

  for (int t = AR_N; t < S_OUT; ++t) {
    float s = nrow[t];                 // independent of y1 -> off critical path
    s = fmaf(y2, c1, s);
    s = fmaf(y3, c2, s);
    s = fmaf(y4, c3, s);
    s = fmaf(y5, c4, s);
    s = fmaf(y6, c5, s);
    s = fmaf(y7, c6, s);
    float y = fmaf(y1, c0, s);         // the only loop-carried fma
    orow[t] = y;
    y7 = y6; y6 = y5; y5 = y4; y4 = y3; y3 = y2; y2 = y1; y1 = y;
  }
}

// ---------------------------------------------------------------------------
// GEMM1: h1 = relu(x @ W1^T + b1).  M=8192, N=128, K=4096.
// 4 waves/block, 16 M-rows per wave, full N=128 in 8 accumulator tiles.
// Two-stage software pipeline: while stage-0 WMMAs execute, stage-1 A/B
// fragments are in flight (manual unroll-by-2 keeps buffers in registers).
// ---------------------------------------------------------------------------
__global__ void __launch_bounds__(128) gemm1_relu_kernel(const float* __restrict__ x,
                                                         const uint16_t* __restrict__ w1b,
                                                         const float* __restrict__ b1,
                                                         uint16_t* __restrict__ h1) {
  const int lane = threadIdx.x & 31;
  const int wave = threadIdx.x >> 5;
  const int half = lane >> 4;
  const int l15  = lane & 15;
  const int m0   = blockIdx.x * 64 + wave * 16;

  const float*    xrow  = x   + (size_t)(m0 + l15) * S_IN + half * 8;
  const uint16_t* wbase = w1b + (size_t)l15 * S_IN + half * 8;

  v8f acc[8];
#pragma unroll
  for (int i = 0; i < 8; ++i) acc[i] = vzero8();

  auto loadA = [&](int k0, float4 (&r)[4]) {
    const float* ap = xrow + k0;
    r[0] = *(const float4*)(ap);
    r[1] = *(const float4*)(ap + 4);
    r[2] = *(const float4*)(ap + 16);
    r[3] = *(const float4*)(ap + 20);
  };
  auto loadB = [&](int k0, ABFrag (&B)[8]) {
#pragma unroll
    for (int nt = 0; nt < 8; ++nt) {
      const uint16_t* wp = wbase + (size_t)nt * (16 * S_IN) + k0;
      B[nt].q[0] = *(const uint4*)(wp);
      B[nt].q[1] = *(const uint4*)(wp + 16);
    }
  };

  float4 Ar0[4], Ar1[4];
  ABFrag Bf0[8], Bf1[8];
  loadA(0, Ar0);
  loadB(0, Bf0);

  for (int k0 = 0; k0 < S_IN; k0 += 64) {
    __builtin_prefetch(xrow + k0 + 128, 0, 1);
    // ---- stage 0: prefetch k0+32, compute on k0 ----
    loadA(k0 + 32, Ar1);
    loadB(k0 + 32, Bf1);
    {
      ABFrag A; pack_a(A, Ar0);
#pragma unroll
      for (int nt = 0; nt < 8; ++nt)
        acc[nt] = __builtin_amdgcn_wmma_f32_16x16x32_bf16(
            false, A.v, false, Bf0[nt].v, (short)0, acc[nt], false, false);
    }
    // ---- stage 1: prefetch k0+64, compute on k0+32 ----
    if (k0 + 64 < S_IN) {
      loadA(k0 + 64, Ar0);
      loadB(k0 + 64, Bf0);
    }
    {
      ABFrag A; pack_a(A, Ar1);
#pragma unroll
      for (int nt = 0; nt < 8; ++nt)
        acc[nt] = __builtin_amdgcn_wmma_f32_16x16x32_bf16(
            false, A.v, false, Bf1[nt].v, (short)0, acc[nt], false, false);
    }
  }

#pragma unroll
  for (int nt = 0; nt < 8; ++nt) {
    const int n = nt * 16 + l15;
    const float bias = b1[n];
#pragma unroll
    for (int r = 0; r < 8; ++r) {
      float v = acc[nt][r] + bias;
      v = v > 0.f ? v : 0.f;
      h1[(size_t)(m0 + half * 8 + r) * H_SZ + n] = f32_to_bf16(v);
    }
  }
}

// ---------------------------------------------------------------------------
// GEMM2: h2 = relu(h1 @ W2^T + b2).  M=8192, N=128, K=128.  A and B both bf16.
// Fully unrolled 4 K-steps with the same two-stage pipeline.
// ---------------------------------------------------------------------------
__global__ void __launch_bounds__(128) gemm2_relu_kernel(const uint16_t* __restrict__ h1,
                                                         const uint16_t* __restrict__ w2b,
                                                         const float* __restrict__ b2,
                                                         uint16_t* __restrict__ h2) {
  const int lane = threadIdx.x & 31;
  const int wave = threadIdx.x >> 5;
  const int half = lane >> 4;
  const int l15  = lane & 15;
  const int m0   = blockIdx.x * 64 + wave * 16;

  const uint16_t* arow  = h1  + (size_t)(m0 + l15) * H_SZ + half * 8;
  const uint16_t* wbase = w2b + (size_t)l15 * H_SZ + half * 8;

  v8f acc[8];
#pragma unroll
  for (int i = 0; i < 8; ++i) acc[i] = vzero8();

  auto loadA = [&](int k0, ABFrag& A) {
    A.q[0] = *(const uint4*)(arow + k0);        // K = half*8 .. +7
    A.q[1] = *(const uint4*)(arow + k0 + 16);   // K = 16+half*8 .. +7
  };
  auto loadB = [&](int k0, ABFrag (&B)[8]) {
#pragma unroll
    for (int nt = 0; nt < 8; ++nt) {
      const uint16_t* wp = wbase + (size_t)nt * (16 * H_SZ) + k0;
      B[nt].q[0] = *(const uint4*)(wp);
      B[nt].q[1] = *(const uint4*)(wp + 16);
    }
  };

  ABFrag A0, A1, Bf0[8], Bf1[8];
  loadA(0, A0);
  loadB(0, Bf0);
#pragma unroll
  for (int ks = 0; ks < 4; ks += 2) {
    loadA((ks + 1) * 32, A1);
    loadB((ks + 1) * 32, Bf1);
#pragma unroll
    for (int nt = 0; nt < 8; ++nt)
      acc[nt] = __builtin_amdgcn_wmma_f32_16x16x32_bf16(
          false, A0.v, false, Bf0[nt].v, (short)0, acc[nt], false, false);
    if (ks + 2 < 4) {
      loadA((ks + 2) * 32, A0);
      loadB((ks + 2) * 32, Bf0);
    }
#pragma unroll
    for (int nt = 0; nt < 8; ++nt)
      acc[nt] = __builtin_amdgcn_wmma_f32_16x16x32_bf16(
          false, A1.v, false, Bf1[nt].v, (short)0, acc[nt], false, false);
  }

#pragma unroll
  for (int nt = 0; nt < 8; ++nt) {
    const int n = nt * 16 + l15;
    const float bias = b2[n];
#pragma unroll
    for (int r = 0; r < 8; ++r) {
      float v = acc[nt][r] + bias;
      v = v > 0.f ? v : 0.f;
      h2[(size_t)(m0 + half * 8 + r) * H_SZ + n] = f32_to_bf16(v);
    }
  }
}

// ---------------------------------------------------------------------------
// GEMM3: out += tanh(h2 @ W3^T + b3)  (out already holds the AR values).
// M=8192, N=4096, K=128. A (16x128 per wave) resident in 32 VGPRs; the
// 256-tile N sweep is unrolled by 2 so next tile's W3 loads overlap the
// current tile's WMMAs + tanh/RMW epilogue.
// ---------------------------------------------------------------------------
__global__ void __launch_bounds__(128) gemm3_tanh_add_kernel(const uint16_t* __restrict__ h2,
                                                             const uint16_t* __restrict__ w3b,
                                                             const float* __restrict__ b3,
                                                             float* __restrict__ out) {
  const int lane = threadIdx.x & 31;
  const int wave = threadIdx.x >> 5;
  const int half = lane >> 4;
  const int l15  = lane & 15;
  const int m0   = blockIdx.x * 64 + wave * 16;

  // Load full-K A fragments once (4 x 32-K slices, 32 VGPRs).
  ABFrag Af[4];
  const uint16_t* arow = h2 + (size_t)(m0 + l15) * H_SZ;
#pragma unroll
  for (int ks = 0; ks < 4; ++ks) {
    const uint16_t* ap = arow + ks * 32 + half * 8;
    Af[ks].q[0] = *(const uint4*)(ap);
    Af[ks].q[1] = *(const uint4*)(ap + 16);
  }

  const uint16_t* wbase = w3b + (size_t)l15 * H_SZ + half * 8;
  auto loadB = [&](int nt, ABFrag (&B)[4]) {
    const uint16_t* wr = wbase + (size_t)nt * (16 * H_SZ);
#pragma unroll
    for (int ks = 0; ks < 4; ++ks) {
      const uint16_t* wp = wr + ks * 32;
      B[ks].q[0] = *(const uint4*)(wp);
      B[ks].q[1] = *(const uint4*)(wp + 16);
    }
  };

  auto tile_epilogue = [&](int nt, const ABFrag (&B)[4]) {
    v8f acc = vzero8();
#pragma unroll
    for (int ks = 0; ks < 4; ++ks)
      acc = __builtin_amdgcn_wmma_f32_16x16x32_bf16(
          false, Af[ks].v, false, B[ks].v, (short)0, acc, false, false);
    const int n = nt * 16 + l15;
    const float bias = b3[n];
    float* orow = out + (size_t)(m0 + half * 8) * S_OUT + n;
#pragma unroll
    for (int r = 0; r < 8; ++r) {
      float* p = orow + (size_t)r * S_OUT;
      *p = fast_tanhf(acc[r] + bias) + *p;   // add AR term already in d_out
    }
  };

  ABFrag Bf0[4], Bf1[4];
  loadB(0, Bf0);
  for (int nt = 0; nt < 256; nt += 2) {
    loadB(nt + 1, Bf1);
    tile_epilogue(nt, Bf0);
    if (nt + 2 < 256) loadB(nt + 2, Bf0);
    tile_epilogue(nt + 1, Bf1);
  }
}

// ---------------------------------------------------------------------------
// Launcher. Input order: x, noise, W1, b1, W2, b2, W3, b3, ar_coef.
// ---------------------------------------------------------------------------
extern "C" void kernel_launch(void* const* d_in, const int* in_sizes, int n_in,
                              void* d_out, int out_size, void* d_ws, size_t ws_size,
                              hipStream_t stream) {
  const float* x     = (const float*)d_in[0];
  const float* noise = (const float*)d_in[1];
  const float* W1    = (const float*)d_in[2];
  const float* b1    = (const float*)d_in[3];
  const float* W2    = (const float*)d_in[4];
  const float* b2    = (const float*)d_in[5];
  const float* W3    = (const float*)d_in[6];
  const float* b3    = (const float*)d_in[7];
  const float* arc   = (const float*)d_in[8];
  float* out = (float*)d_out;

  // Workspace layout (bf16 halves): h1, h2, W1b, W2b, W3b  (~6.3 MB total)
  uint16_t* h1  = (uint16_t*)d_ws;
  uint16_t* h2  = h1  + (size_t)B_SZ * H_SZ;
  uint16_t* w1b = h2  + (size_t)B_SZ * H_SZ;
  uint16_t* w2b = w1b + (size_t)H_SZ * S_IN;
  uint16_t* w3b = w2b + (size_t)H_SZ * H_SZ;

  // Convert weights to bf16 (tiny).
  cvt_bf16_kernel<<<(H_SZ * S_IN + 255) / 256, 256, 0, stream>>>(W1, w1b, H_SZ * S_IN);
  cvt_bf16_kernel<<<(H_SZ * H_SZ + 255) / 256, 256, 0, stream>>>(W2, w2b, H_SZ * H_SZ);
  cvt_bf16_kernel<<<(S_OUT * H_SZ + 255) / 256, 256, 0, stream>>>(W3, w3b, S_OUT * H_SZ);

  // AR recurrence writes pure AR values into d_out (fully overwrites poison).
  ar_filter_kernel<<<B_SZ / 256, 256, 0, stream>>>(noise, arc, out);

  // Fused MLP pipeline (bf16 WMMA, f32 accumulate).
  gemm1_relu_kernel<<<B_SZ / 64, 128, 0, stream>>>(x, w1b, b1, h1);
  gemm2_relu_kernel<<<B_SZ / 64, 128, 0, stream>>>(h1, w2b, b2, h2);
  gemm3_tanh_add_kernel<<<B_SZ / 64, 128, 0, stream>>>(h2, w3b, b3, out);
}
